// GCN_40303973106094
// MI455X (gfx1250) — matrix operand order repaired and checked by reading, not compiled, not run on hardware
//
#include <hip/hip_runtime.h>

#define NPG 24
#define EPG 76
#define NG  8192
#define NN  (NPG*NG)   /* 196608 nodes  */
#define NE  (EPG*NG)   /* 622592 edges  */
#define H   128
#define NODE_F 16
#define EDGE_F 8

typedef __attribute__((ext_vector_type(16))) __bf16 bf16x16;
typedef __attribute__((ext_vector_type(8)))  float  f32x8;

union Frag16 { bf16x16 v; uint4 q[2]; };

__device__ __forceinline__ unsigned short f2bf(float f) {
    union { float f; unsigned u; } x; x.f = f;
    unsigned u = x.u;
    u += 0x7fffu + ((u >> 16) & 1u);          // round-to-nearest-even
    return (unsigned short)(u >> 16);
}

// ---------------- weight f32 -> bf16 conversion ----------------
__global__ void convert_weights_kernel(const float* __restrict__ Wc,
                                       const float* __restrict__ Wr,
                                       unsigned short* __restrict__ Wcb,
                                       unsigned short* __restrict__ Wrb) {
    int i = blockIdx.x * blockDim.x + threadIdx.x;
    const int NC = 4 * H * H;
    if (i < NC) Wcb[i] = f2bf(Wc[i]);
    else { int j = i - NC; if (j < H * H) Wrb[j] = f2bf(Wr[j]); }
}

// ---------------- degrees (with self loop) -> dinv -------------
__global__ void degree_kernel(const int* __restrict__ col,
                              float* __restrict__ dinv) {
    __shared__ int deg[NPG];
    const int g = blockIdx.x, t = threadIdx.x;
    if (t < NPG) deg[t] = 1;                   // self loop
    __syncthreads();
    if (t < EPG) atomicAdd(&deg[col[g * EPG + t] - g * NPG], 1);  // int: deterministic
    __syncthreads();
    if (t < NPG) dinv[g * NPG + t] = rsqrtf((float)deg[t]);
}

// ---------------- node encoder: h = relu(x @ Wn + bn) ----------
__global__ void encoder_kernel(const float* __restrict__ x,
                               const float* __restrict__ Wn,
                               const float* __restrict__ bn,
                               float* __restrict__ h,
                               unsigned short* __restrict__ hbf) {
    const int row = blockIdx.x, t = threadIdx.x;
    const float* xr = x + (size_t)row * NODE_F;
    float acc = bn[t];
#pragma unroll
    for (int k = 0; k < NODE_F; ++k) acc = fmaf(xr[k], Wn[k * H + t], acc);
    acc = fmaxf(acc, 0.f);
    h[(size_t)row * H + t]   = acc;
    hbf[(size_t)row * H + t] = f2bf(acc);
}

// ---- helper: stage the 128x128 bf16 weight matrix into LDS ----
__device__ __forceinline__ void stage_W(const unsigned short* __restrict__ W,
                                        unsigned short* Ws) {
    const uint4* src = (const uint4*)W;   // 2048 x 16B
    uint4* dst = (uint4*)Ws;
#pragma unroll
    for (int i = threadIdx.x; i < (H * H) / 8; i += 256) dst[i] = src[i];
    __syncthreads();
}

// ---- helper: load one A fragment pair (two 16-row tiles) ----
__device__ __forceinline__ void load_A(const unsigned short* A, int row0, int lane,
                                       Frag16 a0[4], Frag16 a1[4]) {
    const int m  = lane & 15;
    const int kh = lane >> 4;
    const unsigned short* Ar0 = A + (size_t)(row0 + m) * H;
    const unsigned short* Ar1 = A + (size_t)(row0 + 16 + m) * H;
#pragma unroll
    for (int kf = 0; kf < 4; ++kf) {
        a0[kf].q[0] = *(const uint4*)(Ar0 + kf * 32 + kh * 8);
        a0[kf].q[1] = *(const uint4*)(Ar0 + kf * 32 + kh * 8 + 16);
        a1[kf].q[0] = *(const uint4*)(Ar1 + kf * 32 + kh * 8);
        a1[kf].q[1] = *(const uint4*)(Ar1 + kf * 32 + kh * 8 + 16);
    }
}

// ---- helper: preload all 4 B fragments of one N-tile from LDS ----
__device__ __forceinline__ void load_B(const unsigned short* Ws, int lane, int nt,
                                       Frag16 b[4]) {
#pragma unroll
    for (int kf = 0; kf < 4; ++kf) {
        const unsigned short* Bcol = Ws + (kf * 32 + (lane >> 4) * 8) * H + nt * 16 + (lane & 15);
        unsigned short* bs = (unsigned short*)b[kf].q;
#pragma unroll
        for (int s = 0; s < 8; ++s) {
            bs[s]     = Bcol[s * H];
            bs[s + 8] = Bcol[(s + 16) * H];
        }
    }
}

// ------- WMMA matmul: C[N,H] = A_bf16 @ W_bf16 + bias ----------
// 256 threads = 8 waves, each wave computes a 32x128 strip (two 16-row tiles).
__global__ void mm_bias_kernel(const unsigned short* __restrict__ A,
                               const unsigned short* __restrict__ W,
                               const float* __restrict__ bias,
                               float* __restrict__ C) {
    __shared__ unsigned short Ws[H * H];
    stage_W(W, Ws);

    const int lane = threadIdx.x & 31;
    const int wave = threadIdx.x >> 5;
    const int row0 = (blockIdx.x * 8 + wave) * 32;

    Frag16 a0[4], a1[4];
    load_A(A, row0, lane, a0, a1);

    const int n  = lane & 15;
    const int mb = (lane >> 4) * 8;
#pragma unroll
    for (int nt = 0; nt < 8; ++nt) {
        Frag16 b[4];
        load_B(Ws, lane, nt, b);
        const float bv = bias[nt * 16 + n];
        f32x8 c0 = { bv, bv, bv, bv, bv, bv, bv, bv };
        f32x8 c1 = c0;
#pragma unroll
        for (int kf = 0; kf < 4; ++kf) {
            c0 = __builtin_amdgcn_wmma_f32_16x16x32_bf16(false, a0[kf].v, false, b[kf].v,
                                                         (short)0, c0, false, false);
            c1 = __builtin_amdgcn_wmma_f32_16x16x32_bf16(false, a1[kf].v, false, b[kf].v,
                                                         (short)0, c1, false, false);
        }
#pragma unroll
        for (int r = 0; r < 8; ++r) {
            C[(size_t)(row0 + mb + r) * H + nt * 16 + n]      = c0[r];
            C[(size_t)(row0 + 16 + mb + r) * H + nt * 16 + n] = c1[r];
        }
    }
}

// --- fused: h = relu(agg + A_bf16 @ Wres + b_res); update bf16 copy ---
__global__ void mm_res_combine_kernel(const unsigned short* Abf,
                                      const unsigned short* __restrict__ W,
                                      const float* __restrict__ bias,
                                      const float* __restrict__ agg,
                                      float* __restrict__ hOut,
                                      unsigned short* hbfOut) {
    __shared__ unsigned short Ws[H * H];
    stage_W(W, Ws);

    const int lane = threadIdx.x & 31;
    const int wave = threadIdx.x >> 5;
    const int row0 = (blockIdx.x * 8 + wave) * 32;

    Frag16 a0[4], a1[4];
    load_A(Abf, row0, lane, a0, a1);

    const int n  = lane & 15;
    const int mb = (lane >> 4) * 8;
#pragma unroll
    for (int nt = 0; nt < 8; ++nt) {
        Frag16 b[4];
        load_B(Ws, lane, nt, b);
        const float bv = bias[nt * 16 + n];
        f32x8 c0 = { bv, bv, bv, bv, bv, bv, bv, bv };
        f32x8 c1 = c0;
#pragma unroll
        for (int kf = 0; kf < 4; ++kf) {
            c0 = __builtin_amdgcn_wmma_f32_16x16x32_bf16(false, a0[kf].v, false, b[kf].v,
                                                         (short)0, c0, false, false);
            c1 = __builtin_amdgcn_wmma_f32_16x16x32_bf16(false, a1[kf].v, false, b[kf].v,
                                                         (short)0, c1, false, false);
        }
#pragma unroll
        for (int r = 0; r < 8; ++r) {
            const size_t i0 = (size_t)(row0 + mb + r) * H + nt * 16 + n;
            const size_t i1 = (size_t)(row0 + 16 + mb + r) * H + nt * 16 + n;
            float v0 = fmaxf(c0[r] + agg[i0], 0.f);
            float v1 = fmaxf(c1[r] + agg[i1], 0.f);
            hOut[i0] = v0;  hbfOut[i0] = f2bf(v0);
            hOut[i1] = v1;  hbfOut[i1] = f2bf(v1);
        }
    }
}

// ---- deterministic per-graph normalized scatter (segment_sum) ----
__global__ void aggregate_kernel(const float* __restrict__ hW,
                                 const int* __restrict__ row,
                                 const int* __restrict__ col,
                                 const float* __restrict__ dinv,
                                 float* __restrict__ agg) {
    __shared__ float hl[NPG][H];
    __shared__ float acc[NPG][H];
    __shared__ float di[NPG];
    const int g = blockIdx.x, t = threadIdx.x;   // t = feature column
    const int nbase = g * NPG;
    if (t < NPG) di[t] = dinv[nbase + t];
    __syncthreads();
#pragma unroll
    for (int v = 0; v < NPG; ++v) {
        float hv = hW[(size_t)(nbase + v) * H + t];
        hl[v][t]  = hv;
        acc[v][t] = hv * di[v] * di[v];          // self-loop term
    }
    const int* rp = row + g * EPG;
    const int* cp = col + g * EPG;
    for (int e = 0; e < EPG; ++e) {              // serial per column -> deterministic
        const int r = rp[e] - nbase;
        const int c = cp[e] - nbase;
        acc[c][t] += hl[r][t] * (di[r] * di[c]);
    }
#pragma unroll
    for (int v = 0; v < NPG; ++v)
        agg[(size_t)(nbase + v) * H + t] = acc[v][t];
}

// ---------------- decoder: one wave per edge -------------------
__global__ void decoder_kernel(const float* __restrict__ h,
                               const float* __restrict__ ea,
                               const int* __restrict__ row,
                               const int* __restrict__ col,
                               const float* __restrict__ We,
                               const float* __restrict__ be,
                               const float* __restrict__ Wd,
                               const float* __restrict__ bd,
                               float* __restrict__ out) {
    const int lane = threadIdx.x & 31;
    const int e = blockIdx.x * 8 + (threadIdx.x >> 5);
    if (e >= NE) return;
    const int r = row[e], c = col[e];
    const int pos = e % EPG;
    const float* ep = ea + (size_t)e * EDGE_F;
    float eav[EDGE_F];
#pragma unroll
    for (int k = 0; k < EDGE_F; ++k) eav[k] = ep[k];
    float partial = 0.f;
#pragma unroll
    for (int j = 0; j < 4; ++j) {
        const int f = lane + 32 * j;
        partial = fmaf(h[(size_t)r * H + f], Wd[f], partial);
        partial = fmaf(h[(size_t)c * H + f], Wd[H + f], partial);
        float t = be[f] + We[(size_t)(EDGE_F + pos) * H + f];   // on-the-fly edge enc
#pragma unroll
        for (int k = 0; k < EDGE_F; ++k) t = fmaf(eav[k], We[k * H + f], t);
        partial = fmaf(fmaxf(t, 0.f), Wd[2 * H + f], partial);
    }
#pragma unroll
    for (int off = 16; off > 0; off >>= 1)
        partial += __shfl_down(partial, off, 32);
    if (lane == 0) out[e] = partial + bd[0];
}

extern "C" void kernel_launch(void* const* d_in, const int* in_sizes, int n_in,
                              void* d_out, int out_size, void* d_ws, size_t ws_size,
                              hipStream_t stream) {
    const float* x  = (const float*)d_in[0];
    const int*   ei = (const int*)d_in[1];
    const float* ea = (const float*)d_in[2];
    /* d_in[3] = batch, unused */
    const float* Wn = (const float*)d_in[4];
    const float* bn = (const float*)d_in[5];
    const float* We = (const float*)d_in[6];
    const float* be = (const float*)d_in[7];
    const float* Wc = (const float*)d_in[8];
    const float* bc = (const float*)d_in[9];
    const float* Wr = (const float*)d_in[10];
    const float* br = (const float*)d_in[11];
    const float* Wd = (const float*)d_in[12];
    const float* bd = (const float*)d_in[13];
    const int* rowI = ei;
    const int* colI = ei + NE;

    char* p = (char*)d_ws;
    float* h    = (float*)p;                   p += (size_t)NN * H * 4;
    float* hW   = (float*)p;                   p += (size_t)NN * H * 4;
    float* agg  = (float*)p;                   p += (size_t)NN * H * 4;
    unsigned short* hbf = (unsigned short*)p;  p += (size_t)NN * H * 2;
    float* dinv = (float*)p;                   p += (size_t)NN * 4;
    unsigned short* Wcb = (unsigned short*)p;  p += (size_t)4 * H * H * 2;
    unsigned short* Wrb = (unsigned short*)p;  p += (size_t)H * H * 2;

    convert_weights_kernel<<<(5 * H * H + 255) / 256, 256, 0, stream>>>(Wc, Wr, Wcb, Wrb);
    degree_kernel<<<NG, 128, 0, stream>>>(colI, dinv);
    encoder_kernel<<<NN, 128, 0, stream>>>(x, Wn, bn, h, hbf);

    for (int l = 0; l < 4; ++l) {
        mm_bias_kernel<<<NN / 256, 256, 0, stream>>>(hbf, Wcb + (size_t)l * H * H,
                                                     bc + l * H, hW);
        aggregate_kernel<<<NG, 128, 0, stream>>>(hW, rowI, colI, dinv, agg);
        mm_res_combine_kernel<<<NN / 256, 256, 0, stream>>>(hbf, Wrb, br, agg, h, hbf);
    }

    decoder_kernel<<<NE / 8, 256, 0, stream>>>(h, ea, rowI, colI, We, be, Wd, bd,
                                               (float*)d_out);
}
